// GRUDecoder_83528523973269
// MI455X (gfx1250) — matrix operand order repaired and checked by reading, not compiled, not run on hardware
//
#include <hip/hip_runtime.h>
#include <hip/hip_bf16.h>

// ---------------------------------------------------------------------------
// GRU decoder for MI455X (gfx1250, wave32, WMMA, async-LDS, persistent scan).
// Pipeline:
//   1) transpose-convert W, U, fc_w  (f32 [K][N] -> bf16 [N][K])
//   2) gather  x_bf16 = bf16(emb[dec_input])            (4096 x 1024)
//   3) GEMM    xp = x @ W + b_i                          (4096 x 3072, f32)
//   4) ONE persistent kernel: 256 GRU steps, U in LDS, h in registers,
//      device-wide atomic barrier between steps
//   5) GEMM    logits = outs_bf16 @ fc_w + fc_b          (4096 x 32000)
// ---------------------------------------------------------------------------

typedef __attribute__((ext_vector_type(16))) __bf16 v16bf;
typedef __attribute__((ext_vector_type(8)))  __bf16 v8bf;
typedef __attribute__((ext_vector_type(8)))  float  v8f;

union V16 {
    v16bf v;
    v8bf  h[2];
};

__device__ __forceinline__ v8f wmma_bf16(v16bf a, v16bf b, v8f c) {
    // emits v_wmma_f32_16x16x32_bf16
    return __builtin_amdgcn_wmma_f32_16x16x32_bf16(false, a, false, b,
                                                   (short)0, c, false, false);
}

// CDNA5 async global->LDS copy (tracked by ASYNCcnt, bypasses VGPRs).
// VDST = 32-bit LDS byte address, VADDR = 64-bit global address.
__device__ __forceinline__ void async_copy_b128(void* lds_dst, const void* gsrc) {
    const unsigned lds_addr = (unsigned)(unsigned long long)lds_dst; // low 32b = LDS offset
    asm volatile("global_load_async_to_lds_b128 %0, %1, off"
                 :: "v"(lds_addr), "v"(gsrc)
                 : "memory");
}
__device__ __forceinline__ void wait_async0() {
    asm volatile("s_wait_asynccnt 0x0" ::: "memory");
}

__device__ __forceinline__ float sigmoidf_(float x) {
    return 1.0f / (1.0f + __expf(-x));
}

// Device-wide phase barrier: one add per block, all lanes poll (same address
// -> uniform wave exit, EXEC back to all-ones before the next WMMA chain).
__device__ __forceinline__ void gsync(unsigned* cnt, unsigned target) {
    __threadfence();
    if (threadIdx.x == 0)
        __hip_atomic_fetch_add(cnt, 1u, __ATOMIC_RELEASE, __HIP_MEMORY_SCOPE_AGENT);
    while (__hip_atomic_load(cnt, __ATOMIC_ACQUIRE, __HIP_MEMORY_SCOPE_AGENT) < target)
        __builtin_amdgcn_s_sleep(1);
    __threadfence();
}

static constexpr int Bsz = 16;
static constexpr int T   = 256;
static constexpr int H   = 1024;
static constexpr int V   = 32000;

// ---------------------------------------------------------------------------
// Transpose + convert: in f32 [R][C] (row-major) -> out bf16 [C][R]
// Grid: (C/32, R/32), block 256 (= 32x8)
// ---------------------------------------------------------------------------
__global__ __launch_bounds__(256)
void k_transpose_cvt(const float* __restrict__ in, __bf16* __restrict__ out,
                     int R, int C) {
    __shared__ float tile[32][33];
    const int tx = threadIdx.x & 31;
    const int ty = threadIdx.x >> 5;
    const int r0 = blockIdx.y * 32;
    const int c0 = blockIdx.x * 32;
#pragma unroll
    for (int i = 0; i < 4; ++i)
        tile[ty + i * 8][tx] = in[(size_t)(r0 + ty + i * 8) * C + c0 + tx];
    __syncthreads();
#pragma unroll
    for (int i = 0; i < 4; ++i)
        out[(size_t)(c0 + ty + i * 8) * R + r0 + tx] = (__bf16)tile[tx][ty + i * 8];
}

// ---------------------------------------------------------------------------
// Embedding gather + f32->bf16.  One block per (b,t) row.
// ---------------------------------------------------------------------------
__global__ __launch_bounds__(256)
void k_gather_emb(const int* __restrict__ dec_input,
                  const float* __restrict__ emb,
                  __bf16* __restrict__ xb) {
    const int row = blockIdx.x;                 // b*T + t
    const int tok = dec_input[row];
    const float* src = emb + (size_t)tok * H;
    __bf16* dst = xb + (size_t)row * H;
    const int k = threadIdx.x * 4;              // 256 threads * 4 = 1024
    float4 f = *(const float4*)(src + k);
    dst[k + 0] = (__bf16)f.x;
    dst[k + 1] = (__bf16)f.y;
    dst[k + 2] = (__bf16)f.z;
    dst[k + 3] = (__bf16)f.w;
}

// ---------------------------------------------------------------------------
// WMMA GEMM:  C[M][N] = A[M][K] (bf16, row-major) * Bt[N][K]^T (bf16) + bias
// Block 256 threads = 8 waves as 2(M) x 4(N); block tile 64x128,
// wave tile 32x32 (2x2 WMMA 16x16x32 accumulators).
// A double-buffered in LDS via async copies; B software-pipelined in regs.
// Epilogue stages the f32 tile in LDS for coalesced float4 stores.
// ---------------------------------------------------------------------------
__global__ __launch_bounds__(256)
void k_gemm_bf16(const __bf16* __restrict__ A, const __bf16* __restrict__ Bt,
                 const float* __restrict__ bias, float* __restrict__ C,
                 int M, int N, int K) {
    __shared__ __bf16 As[2][64][32];            // 8 KB (double buffer)
    __shared__ float  Cs[64][128];              // 32 KB (store staging)

    const int lane = threadIdx.x & 31;
    const int wv   = threadIdx.x >> 5;
    const int wm   = wv >> 2;                   // 0..1
    const int wn   = wv & 3;                    // 0..3
    const int m0   = blockIdx.y * 64;
    const int n0   = blockIdx.x * 128;
    const int lh   = (lane >> 4) * 8;           // unused; A loads also use kb
    const int kb   = (lane >> 4) * 16;          // B half-K offset per lane group
    const int ln   = lane & 15;

    const int idx  = threadIdx.x * 8;           // per-thread A-copy slice
    const int arow = idx >> 5;
    const int akol = idx & 31;
    const __bf16* aSrc = &A[(size_t)(m0 + arow) * K + akol];

    const __bf16* bSrc[2];
#pragma unroll
    for (int j = 0; j < 2; ++j)
        bSrc[j] = &Bt[(size_t)(n0 + wn * 32 + j * 16 + ln) * K + kb];

    // ---- prologue ----
    async_copy_b128(&As[0][arow][akol], aSrc);
    V16 bcur[2];
#pragma unroll
    for (int j = 0; j < 2; ++j)
        bcur[j].v = *(const v16bf*)(bSrc[j]);
    wait_async0();
    __syncthreads();

    v8f acc[2][2] = {};

    for (int k0 = 0; k0 < K; k0 += 32) {
        const int  buf     = (k0 >> 5) & 1;
        const bool notlast = (k0 + 32) < K;

        if (notlast)
            async_copy_b128(&As[buf ^ 1][arow][akol], aSrc + k0 + 32);

#pragma unroll
        for (int j = 0; j < 2; ++j)
            __builtin_prefetch(bSrc[j] + k0 + 256, 0, 3);

        V16 bnext[2];
        if (notlast) {
#pragma unroll
            for (int j = 0; j < 2; ++j)
                bnext[j].v = *(const v16bf*)(bSrc[j] + k0 + 32);
        }

        V16 a[2];
#pragma unroll
        for (int i = 0; i < 2; ++i) {
            const int r = wm * 32 + i * 16 + ln;
            a[i].h[0] = *(const v8bf*)&As[buf][r][kb];
            a[i].h[1] = *(const v8bf*)&As[buf][r][kb + 8];
        }

#pragma unroll
        for (int i = 0; i < 2; ++i)
#pragma unroll
            for (int j = 0; j < 2; ++j)
                acc[i][j] = wmma_bf16(a[i].v, bcur[j].v, acc[i][j]);

        bcur[0] = bnext[0];
        bcur[1] = bnext[1];

        if (notlast) {
            wait_async0();
            __syncthreads();
        }
    }

    // ---- epilogue: stage in LDS, coalesced float4 stores ----
    __syncthreads();
#pragma unroll
    for (int i = 0; i < 2; ++i)
#pragma unroll
        for (int j = 0; j < 2; ++j)
#pragma unroll
            for (int v = 0; v < 8; ++v)
                Cs[wm * 32 + i * 16 + v + ((lane >> 4) * 8)]
                  [wn * 32 + j * 16 + ln] = acc[i][j][v];
    __syncthreads();

    const int r  = threadIdx.x >> 2;
    const int c0 = (threadIdx.x & 3) * 32;
#pragma unroll
    for (int c = 0; c < 32; c += 4) {
        float4 vv = *(float4*)&Cs[r][c0 + c];
        float4 bv = *(const float4*)&bias[n0 + c0 + c];
        vv.x += bv.x; vv.y += bv.y; vv.z += bv.z; vv.w += bv.w;
        *(float4*)&C[(size_t)(m0 + r) * N + n0 + c0 + c] = vv;
    }
}

// ---------------------------------------------------------------------------
// Persistent GRU scan: 64 blocks x 32 threads (1 wave), block owns 16 H-cols.
// U slice (3 gates x 16 cols x 1024 K bf16 = 96 KB) preloaded once into LDS
// via async copies; h kept in f32 registers for the owned (batch,col) pairs;
// cross-block h exchanged as bf16 (ping/pong); device-wide barrier per step.
// ---------------------------------------------------------------------------
__global__ __launch_bounds__(32)
void k_gru_scan(const float* __restrict__ dec_hidden,
                const float* __restrict__ xp,
                const __bf16* __restrict__ Ut,
                const float* __restrict__ b_r,
                const int* __restrict__ dec_input,
                __bf16* __restrict__ outs_bf,
                __bf16* __restrict__ hbf0, __bf16* __restrict__ hbf1,
                float* __restrict__ h_final,
                unsigned* __restrict__ sync_cnt) {
    __shared__ __bf16 Us[3][16][H];             // 96 KB, resident all 256 steps

    const int lane = threadIdx.x & 31;
    const int ln   = lane & 15;
    const int hi   = lane >> 4;                 // 0/1 half-wave
    const int lh   = hi * 8;
    const int kb   = hi * 16;
    const int cb   = blockIdx.x * 16;           // owned H-column base
    const int c    = cb + ln;
    const unsigned nb = gridDim.x;

    // ---- one-time async preload of the U slice (48 rows x 2 KB) ----
    for (int ch = threadIdx.x; ch < 48 * 128; ch += 32) {
        const int r  = ch >> 7;                 // 0..47
        const int o  = (ch & 127) * 8;          // bf16 offset within row
        const int g  = r >> 4;
        const int cl = r & 15;
        async_copy_b128(&Us[g][cl][o], &Ut[(size_t)(g * H + cb + cl) * H + o]);
    }

    // ---- init owned h in registers; publish bf16 h into buf0 ----
    float hreg[8], oprev[8];
#pragma unroll
    for (int v = 0; v < 8; ++v) {
        const int bb = v + hi * 8;
        hreg[v]  = dec_hidden[bb * H + c];
        oprev[v] = 0.0f;
        hbf0[bb * H + c] = (__bf16)hreg[v];
    }
    const float brz = b_r[c];
    const float brr = b_r[H + c];
    const float brh = b_r[2 * H + c];

    wait_async0();
    gsync(sync_cnt, nb * 1);                    // phase 0: h_bf(0) visible

    for (int t = 0; t < T; ++t) {
        const __bf16* hin  = (t & 1) ? hbf1 : hbf0;
        __bf16*       hout = (t & 1) ? hbf0 : hbf1;

        // hoisted per-step loads (latency hidden behind the K loop)
        float xz[8], xr[8], xh[8];
        int   mk[8];
#pragma unroll
        for (int v = 0; v < 8; ++v) {
            const int bb = v + hi * 8;
            const size_t xbase = ((size_t)bb * T + t) * (3 * H);
            xz[v] = xp[xbase + c];
            xr[v] = xp[xbase + H + c];
            xh[v] = xp[xbase + 2 * H + c];
            mk[v] = dec_input[bb * T + t];
        }

        // K loop: A from global bf16 h (depth-2 pipeline), B from LDS
        v8f az = {}, ar = {}, ah = {};
        const __bf16* arow = &hin[ln * H];
        V16 a0, a1;
        a0.h[0] = *(const v8bf*)(arow + kb);
        a0.h[1] = *(const v8bf*)(arow + kb + 8);
        a1.h[0] = *(const v8bf*)(arow + 32 + kb);
        a1.h[1] = *(const v8bf*)(arow + 40 + kb);

        for (int k0 = 0; k0 < H; k0 += 32) {
            V16 an;
            if (k0 + 64 < H) {
                an.h[0] = *(const v8bf*)(arow + k0 + 64 + kb);
                an.h[1] = *(const v8bf*)(arow + k0 + 72 + kb);
            }
            V16 bz, br2, bh2;
            bz.v  = *(const v16bf*)&Us[0][ln][k0 + kb];
            br2.v = *(const v16bf*)&Us[1][ln][k0 + kb];
            bh2.v = *(const v16bf*)&Us[2][ln][k0 + kb];

            az = wmma_bf16(a0.v, bz.v,  az);
            ar = wmma_bf16(a0.v, br2.v, ar);
            ah = wmma_bf16(a0.v, bh2.v, ah);

            a0 = a1;
            a1 = an;
        }

        // gate math + masked update, all in registers
#pragma unroll
        for (int v = 0; v < 8; ++v) {
            const int bb = v + hi * 8;
            const float z  = sigmoidf_(xz[v] + az[v] + brz);
            const float r  = sigmoidf_(xr[v] + ar[v] + brr);
            const float hh = tanhf(xh[v] + r * (ah[v] + brh));
            const float hnew = z * hreg[v] + (1.0f - z) * hh;
            const bool  m  = mk[v] != 0;
            hreg[v]  = m ? hnew : hreg[v];
            oprev[v] = m ? hnew : oprev[v];
            hout[bb * H + c] = (__bf16)hreg[v];
            outs_bf[((size_t)bb * T + t) * H + c] = (__bf16)oprev[v];
        }

        if (t + 1 < T)
            gsync(sync_cnt, nb * (unsigned)(t + 2));
    }

    // h_final straight to the output buffer
#pragma unroll
    for (int v = 0; v < 8; ++v) {
        const int bb = v + hi * 8;
        h_final[bb * H + c] = hreg[v];
    }
}

// ---------------------------------------------------------------------------
extern "C" void kernel_launch(void* const* d_in, const int* in_sizes, int n_in,
                              void* d_out, int out_size, void* d_ws, size_t ws_size,
                              hipStream_t stream) {
    (void)in_sizes; (void)n_in; (void)out_size; (void)ws_size;

    const int*   dec_input  = (const int*)  d_in[0];   // (B,T) int32
    const float* dec_hidden = (const float*)d_in[1];   // (B,H)
    const float* emb        = (const float*)d_in[2];   // (EV,H)
    const float* W          = (const float*)d_in[3];   // (H,3H)
    const float* U          = (const float*)d_in[4];   // (H,3H)
    const float* b_i        = (const float*)d_in[5];   // (3H,)
    const float* b_r        = (const float*)d_in[6];   // (3H,)
    const float* fc_w       = (const float*)d_in[7];   // (H,V)
    const float* fc_b       = (const float*)d_in[8];   // (V,)

    float* out = (float*)d_out;                        // logits ++ h_final

    // ---- workspace carve-up (256B-aligned) ----
    char* ws = (char*)d_ws;
    size_t off = 0;
    auto carve = [&](size_t bytes) { char* p = ws + off; off += (bytes + 255) & ~(size_t)255; return p; };
    __bf16*   x_bf    = (__bf16*)  carve((size_t)Bsz * T * H * 2);        // 8.4 MB
    __bf16*   Wt_bf   = (__bf16*)  carve((size_t)3 * H * H * 2);          // 6.3 MB
    __bf16*   Ut_bf   = (__bf16*)  carve((size_t)3 * H * H * 2);          // 6.3 MB
    __bf16*   fcwT_bf = (__bf16*)  carve((size_t)V * H * 2);              // 65.5 MB
    float*    xp      = (float*)   carve((size_t)Bsz * T * 3 * H * 4);    // 50.3 MB
    __bf16*   outs_bf = (__bf16*)  carve((size_t)Bsz * T * H * 2);        // 8.4 MB
    __bf16*   hbf0    = (__bf16*)  carve((size_t)Bsz * H * 2);
    __bf16*   hbf1    = (__bf16*)  carve((size_t)Bsz * H * 2);
    unsigned* scnt    = (unsigned*)carve(256);

    // ---- 1) transpose-convert weights to bf16 [N][K] ----
    k_transpose_cvt<<<dim3(3 * H / 32, H / 32), 256, 0, stream>>>(W,    Wt_bf,   H, 3 * H);
    k_transpose_cvt<<<dim3(3 * H / 32, H / 32), 256, 0, stream>>>(U,    Ut_bf,   H, 3 * H);
    k_transpose_cvt<<<dim3(V / 32,     H / 32), 256, 0, stream>>>(fc_w, fcwT_bf, H, V);

    // ---- 2) embedding gather -> bf16 ----
    k_gather_emb<<<Bsz * T, 256, 0, stream>>>(dec_input, emb, x_bf);

    // ---- 3) xp = x @ W + b_i ----
    k_gemm_bf16<<<dim3(3 * H / 128, Bsz * T / 64), 256, 0, stream>>>(
        x_bf, Wt_bf, b_i, xp, Bsz * T, 3 * H, H);

    // ---- 4) persistent GRU scan (single launch, 256 steps on-chip) ----
    hipMemsetAsync(scnt, 0, 256, stream);       // reset phase barrier
    k_gru_scan<<<H / 16, 32, 0, stream>>>(dec_hidden, xp, Ut_bf, b_r,
                                          dec_input, outs_bf, hbf0, hbf1,
                                          out + (size_t)Bsz * T * V, scnt);

    // ---- 5) logits = outs @ fc_w + fc_b ----
    k_gemm_bf16<<<dim3(V / 128, Bsz * T / 64), 256, 0, stream>>>(
        outs_bf, fcwT_bf, fc_b, out, Bsz * T, V, H);
}
